// StaticRNN_51384988729861
// MI455X (gfx1250) — matrix operand order repaired and checked by reading, not compiled, not run on hardware
//
#include <hip/hip_runtime.h>
#include <hip/hip_bf16.h>

// LSTM static RNN: B=128, T=512, D=512, H=512, gates i,j,f,o, forget bias 1.0
//
// Phase 0: pack W (f32 [1024,2048]) into bf16 WMMA B-fragment layout (Wx / Wh).
// Phase 1: Zx[t*B+b, 4H] = x @ Wx + b   -- parallel bf16-WMMA GEMM (f32 accum),
//          2 M-tiles per wave so each B fragment feeds 2 WMMAs from registers.
// Phase 2: per t (512 launches): Zh = h_{t-1} @ Wh via WMMA while the Zx tile
//          streams into LDS via global_load_async_to_lds_b128; f32 gate math,
//          masked state update, outputs[t] written.
// h is ping-ponged in bf16 for the next step's A operand; c/h state stays f32.

#define B_   128
#define T_   512
#define D_   512
#define H_   512
#define NG   2048            // 4*H
#define KST  16              // K-steps per 512-deep GEMM (512 / 32)

typedef __attribute__((ext_vector_type(16))) __bf16 v16bf;
typedef __attribute__((ext_vector_type(8)))  __bf16 v8bf;
typedef __attribute__((ext_vector_type(8)))  float  v8f;

static __device__ inline v8f wmma_bf16(v16bf a, v16bf b, v8f c) {
  // D(f32 16x16) = A(bf16 16x32) * B(bf16 32x16) + C
  return __builtin_amdgcn_wmma_f32_16x16x32_bf16(false, a, false, b, (short)0, c,
                                                 false, false);
}

// ---------------------------------------------------------------------------
// B-fragment packed layout: frag = nTile*KST + kstep, 512 bf16 per fragment,
// lane-major: lane holds N = nTile*16 + (lane&15),
//             K = kstep*32 + (j<8 ? j : 8+j) + (lane>=16 ? 8 : 0), j=0..15  (contiguous 32B)
// ---------------------------------------------------------------------------
static __device__ inline v16bf load_b_frag(const __bf16* __restrict__ P, int frag) {
  int lane = threadIdx.x & 31;
  return *(const v16bf*)(P + ((size_t)frag << 9) + ((size_t)lane << 4));
}

// A fragment (16x32 bf16) from f32 source X laid out [B][T][D]; logical GEMM
// row m = t*B + b.  Lane layout per ISA: lanes 0-15 rows M=0..15 carrying
// K = k0+{0..7, 16..23}; lanes 16-31 same rows carrying K = k0+{8..15, 24..31}.
static __device__ inline v16bf load_a_from_x(const float* __restrict__ X, int m0, int k0) {
  int lane = threadIdx.x & 31;
  int row  = m0 + (lane & 15);
  int t    = row >> 7;           // row / B_
  int b    = row & 127;          // row % B_
  const float* p = X + ((size_t)b * T_ + t) * D_ + k0 + ((lane >> 4) << 3);
  v16bf a;
#pragma unroll
  for (int j = 0; j < 8; ++j) {
    a[j]     = (__bf16)p[j];
    a[8 + j] = (__bf16)p[16 + j];
  }
  return a;
}

// A fragment from bf16 h-state [B][H]
static __device__ inline v16bf load_a_from_h(const __bf16* __restrict__ Hm, int m0, int k0) {
  int lane = threadIdx.x & 31;
  int row  = m0 + (lane & 15);
  const __bf16* p = Hm + (size_t)row * H_ + k0 + ((lane >> 4) << 3);
  v8bf lo = *(const v8bf*)p;
  v8bf hi = *(const v8bf*)(p + 16);
  v16bf a;
#pragma unroll
  for (int j = 0; j < 8; ++j) { a[j] = lo[j]; a[8 + j] = hi[j]; }
  return a;
}

// ---------------------------------------------------------------------------
// Phase 0: pack W rows [0,512) -> Wx fragments, rows [512,1024) -> Wh fragments
// ---------------------------------------------------------------------------
__global__ void pack_w_kernel(const float* __restrict__ W,
                              __bf16* __restrict__ wxp, __bf16* __restrict__ whp) {
  const int HALF = 512 * NG;                       // elements per packed half
  int idx = blockIdx.x * blockDim.x + threadIdx.x;
  if (idx >= 2 * HALF) return;
  __bf16* dst; int kOff; int e = idx;
  if (e < HALF) { dst = wxp; kOff = 0; }
  else          { dst = whp; kOff = 512; e -= HALF; }
  int frag  = e >> 9;
  int rem   = e & 511;
  int lane  = rem >> 4;
  int j     = rem & 15;
  int nTile = frag >> 4;         // frag / KST
  int kstep = frag & 15;         // frag % KST
  int n = nTile * 16 + (lane & 15);
  int k = kstep * 32 + ((lane >> 4) << 3) + (j < 8 ? j : 8 + j);
  dst[e] = (__bf16)W[(size_t)(kOff + k) * NG + n];
}

// ---------------------------------------------------------------------------
// Init: copy init_c / init_h into state buffers (+ bf16 copy of h)
// ---------------------------------------------------------------------------
__global__ void init_state_kernel(const float* __restrict__ ic, const float* __restrict__ ih,
                                  float* __restrict__ c_state, float* __restrict__ h_state,
                                  __bf16* __restrict__ h_bf) {
  int i = blockIdx.x * blockDim.x + threadIdx.x;
  if (i < B_ * H_) {
    c_state[i] = ic[i];
    float h = ih[i];
    h_state[i] = h;
    h_bf[i] = (__bf16)h;
  }
}

// ---------------------------------------------------------------------------
// Phase 1: Zx[m, 0:2048] = X @ Wx + bias,  M = T*B, K = 512, N = 2048
// block = 128 threads (4 waves); block tile = 32 rows x 256 cols.
// Each wave owns 2 M-tiles x 4 N-tiles: every B fragment is reused by 2 WMMAs.
// ---------------------------------------------------------------------------
__global__ __launch_bounds__(128) void gemm_x_kernel(const float* __restrict__ X,
                                                     const float* __restrict__ bias,
                                                     const __bf16* __restrict__ wxp,
                                                     float* __restrict__ Zx) {
  int wave  = threadIdx.x >> 5;
  int lane  = threadIdx.x & 31;
  int m0    = blockIdx.x * 32;
  int nbase = blockIdx.y * 256 + wave * 64;
  v8f acc[2][4] = {};
#pragma unroll 2
  for (int ks = 0; ks < KST; ++ks) {
    v16bf a0 = load_a_from_x(X, m0,      ks * 32);
    v16bf a1 = load_a_from_x(X, m0 + 16, ks * 32);
#pragma unroll
    for (int s = 0; s < 4; ++s) {
      v16bf bb = load_b_frag(wxp, ((nbase >> 4) + s) * KST + ks);
      acc[0][s] = wmma_bf16(a0, bb, acc[0][s]);
      acc[1][s] = wmma_bf16(a1, bb, acc[1][s]);
    }
  }
  // C/D layout: element (vgpr r, lane) -> row mt0 + r + 8*(lane>=16), col (lane&15)
#pragma unroll
  for (int mt = 0; mt < 2; ++mt) {
    int rbase = m0 + mt * 16 + ((lane >> 4) << 3);
#pragma unroll
    for (int s = 0; s < 4; ++s) {
      int col  = nbase + s * 16 + (lane & 15);
      float bv = bias[col];
#pragma unroll
      for (int r = 0; r < 8; ++r)
        Zx[(size_t)(rbase + r) * NG + col] = acc[mt][s][r] + bv;
    }
  }
}

// ---------------------------------------------------------------------------
// Phase 2: one timestep. grid (B/16, H/64), 128 threads; wave g = gate g.
// The block's Zx tile (4 gates x 16 rows x 64 cols, 16KB) is streamed into
// LDS with async loads overlapping the WMMA loop; a single s_wait_asynccnt 0
// + barrier orders it before the pointwise stage.
// ---------------------------------------------------------------------------
__global__ __launch_bounds__(128) void lstm_step_kernel(
    const float* __restrict__ Zx, const __bf16* __restrict__ whp,
    const int* __restrict__ num_words,
    float* __restrict__ c_state, float* __restrict__ h_state,
    const __bf16* __restrict__ h_in, __bf16* __restrict__ h_out,
    float* __restrict__ out, int t) {
  __shared__ float zg[4][16][64];    // post-GEMM gate slabs (cross-wave exchange)
  __shared__ float zxt[4][16][64];   // async-staged Zx tile
  int g    = threadIdx.x >> 5;       // gate index: 0=i 1=j 2=f 3=o
  int lane = threadIdx.x & 31;
  int m0   = blockIdx.x * 16;        // batch tile
  int h0   = blockIdx.y * 64;        // hidden-column group

  // Kick off the async Zx tile copy: 1024 x 16B chunks, 8 per thread.
  // chunk q -> gate gq = q>>8, row rq = (q>>4)&15, col cq = (q&15)*4
  {
    size_t rowbase = (size_t)t * B_ + m0;
#pragma unroll
    for (int i = 0; i < 8; ++i) {
      int q  = threadIdx.x + i * 128;
      int gq = q >> 8;
      int rq = (q >> 4) & 15;
      int cq = (q & 15) << 2;
      const float* gptr = Zx + (rowbase + rq) * NG + gq * H_ + h0 + cq;
      unsigned lptr = (unsigned)(unsigned long long)(const void*)&zxt[gq][rq][cq];
      asm volatile("global_load_async_to_lds_b128 %0, %1, off"
                   :: "v"(lptr), "v"(gptr) : "memory");
    }
  }

  v8f acc[4] = {};
#pragma unroll 2
  for (int ks = 0; ks < KST; ++ks) {
    v16bf a = load_a_from_h(h_in, m0, ks * 32);
#pragma unroll
    for (int s = 0; s < 4; ++s) {
      int nt = ((g * H_ + h0) >> 4) + s;
      v16bf bb = load_b_frag(whp, nt * KST + ks);
      acc[s] = wmma_bf16(a, bb, acc[s]);
    }
  }

  // Wait for this wave's async chunks; barrier makes the whole tile visible.
  asm volatile("s_wait_asynccnt 0" ::: "memory");
  __syncthreads();

  // z = Zh + Zx -> LDS (per-gate slab)
  int rHalf = (lane >> 4) << 3;
#pragma unroll
  for (int s = 0; s < 4; ++s) {
    int cl = s * 16 + (lane & 15);
#pragma unroll
    for (int r = 0; r < 8; ++r) {
      int rl = rHalf + r;
      zg[g][rl][cl] = acc[s][r] + zxt[g][rl][cl];
    }
  }
  __syncthreads();

  // pointwise LSTM update: 16x64 = 1024 elements, 8 per thread
#pragma unroll
  for (int i = 0; i < 8; ++i) {
    int e  = threadIdx.x + i * 128;
    int rr = e >> 6, cc = e & 63;
    int b  = m0 + rr, hc = h0 + cc;
    float iG = zg[0][rr][cc], jG = zg[1][rr][cc];
    float fG = zg[2][rr][cc], oG = zg[3][rr][cc];
    size_t sidx = (size_t)b * H_ + hc;
    float c_old = c_state[sidx];
    float h_old = h_state[sidx];
    float fs = 1.f / (1.f + __expf(-(fG + 1.0f)));   // forget bias 1.0
    float is = 1.f / (1.f + __expf(-iG));
    float os = 1.f / (1.f + __expf(-oG));
    float nc = c_old * fs + is * tanhf(jG);
    float nh = tanhf(nc) * os;
    bool mask = t < num_words[b];
    float c2 = mask ? nc : c_old;
    float h2 = mask ? nh : h_old;
    c_state[sidx] = c2;
    h_state[sidx] = h2;
    h_out[sidx]   = (__bf16)h2;
    out[((size_t)t * B_ + b) * H_ + hc] = mask ? nh : 0.f;
  }
}

// Copy final c/h into the output tail (tuple return: outputs, final_c, final_h)
__global__ void finalize_kernel(const float* __restrict__ c_state,
                                const float* __restrict__ h_state,
                                float* __restrict__ out_tail) {
  int i = blockIdx.x * blockDim.x + threadIdx.x;
  if (i < B_ * H_) {
    out_tail[i]            = c_state[i];
    out_tail[B_ * H_ + i]  = h_state[i];
  }
}

// ---------------------------------------------------------------------------
extern "C" void kernel_launch(void* const* d_in, const int* in_sizes, int n_in,
                              void* d_out, int out_size, void* d_ws, size_t ws_size,
                              hipStream_t stream) {
  (void)in_sizes; (void)n_in; (void)out_size; (void)ws_size;
  const float* X      = (const float*)d_in[0];   // [B, T, D]
  const int*   nwords = (const int*)d_in[1];     // [B]
  const float* init_c = (const float*)d_in[2];   // [B, H]
  const float* init_h = (const float*)d_in[3];   // [B, H]
  const float* W      = (const float*)d_in[4];   // [D+H, 4H]
  const float* bias   = (const float*)d_in[5];   // [4H]
  float* out          = (float*)d_out;

  // workspace carve-up (256B aligned)
  char* ws = (char*)d_ws;
  size_t off = 0;
  auto carve = [&](size_t bytes) -> void* {
    void* p = ws + off;
    off = (off + bytes + 255) & ~(size_t)255;
    return p;
  };
  __bf16* wxp     = (__bf16*)carve((size_t)512 * NG * sizeof(__bf16));  // 2 MB
  __bf16* whp     = (__bf16*)carve((size_t)512 * NG * sizeof(__bf16));  // 2 MB
  __bf16* hbuf0   = (__bf16*)carve((size_t)B_ * H_ * sizeof(__bf16));
  __bf16* hbuf1   = (__bf16*)carve((size_t)B_ * H_ * sizeof(__bf16));
  float*  c_state = (float*)carve((size_t)B_ * H_ * sizeof(float));
  float*  h_state = (float*)carve((size_t)B_ * H_ * sizeof(float));
  float*  Zx      = (float*)carve((size_t)T_ * B_ * NG * sizeof(float)); // 512 MB

  // Phase 0: pack weights into WMMA fragment layout
  pack_w_kernel<<<(2 * 512 * NG) / 256, 256, 0, stream>>>(W, wxp, whp);
  // state init
  init_state_kernel<<<(B_ * H_ + 255) / 256, 256, 0, stream>>>(
      init_c, init_h, c_state, h_state, hbuf0);
  // Phase 1: parallel input GEMM  (M/32 x N/256 blocks)
  gemm_x_kernel<<<dim3((T_ * B_) / 32, NG / 256), 128, 0, stream>>>(X, bias, wxp, Zx);
  // Phase 2: sequential recurrence, ping-pong bf16 h
  __bf16* hb[2] = {hbuf0, hbuf1};
  for (int t = 0; t < T_; ++t) {
    lstm_step_kernel<<<dim3(B_ / 16, H_ / 64), 128, 0, stream>>>(
        Zx, whp, nwords, c_state, h_state, hb[t & 1], hb[(t + 1) & 1], out, t);
  }
  finalize_kernel<<<(B_ * H_ + 255) / 256, 256, 0, stream>>>(
      c_state, h_state, out + (size_t)T_ * B_ * H_);
}
